// EDSR_11819749998894
// MI455X (gfx1250) — 8-bit matrix operand order repaired and checked by reading, compiled, not run on hardware
//
#include <hip/hip_runtime.h>
#include <cstdint>
#include <cstddef>


typedef int v8i __attribute__((ext_vector_type(8)));
typedef unsigned int uint32;

#define NRES   16
#define NF     64
#define HW     256
#define NIMG   8
#define OUTHW  512
#define KCONST 130.0f

#define TW 32
#define TH 32
#define HTW (TW + 2)
#define HTH (TH + 2)
#define LPAD 72
#define ACT_BYTES (HTH * HTW * LPAD)
#define LDS_BYTES (ACT_BYTES + NF * 4)

static constexpr size_t FEAT_ELEMS = (size_t)NIMG * NF * HW * HW;
static constexpr size_t FEAT_BYTES = FEAT_ELEMS * sizeof(float);

__device__ __forceinline__ float rgb_mean(int c) {
    return c == 0 ? 255.0f * 0.4488f : (c == 1 ? 255.0f * 0.4371f : 255.0f * 0.4040f);
}

__global__ void prep_weights_kernel(const float* __restrict__ rb_w,
                                    int8_t* __restrict__ bsign,
                                    float* __restrict__ bscale) {
    int o = blockIdx.x * blockDim.x + threadIdx.x;
    if (o >= NRES * NF) return;
    const float* w = rb_w + (size_t)o * NF * 9;
    float s = 0.0f;
    for (int i = 0; i < NF * 9; ++i) s += fabsf(w[i]);
    bscale[o] = s * (1.0f / (NF * 9));
    int8_t* bs = bsign + (size_t)o * 9 * NF;
    for (int ic = 0; ic < NF; ++ic)
        for (int t = 0; t < 9; ++t) {
            float v = w[ic * 9 + t];
            bs[t * NF + ic] = (v > 0.0f) ? 1 : ((v < 0.0f) ? -1 : 0);
        }
}

__global__ void head_kernel(const float* __restrict__ x,
                            const float* __restrict__ hw_,
                            const float* __restrict__ hb,
                            float* __restrict__ feat) {
    size_t idx = (size_t)blockIdx.x * blockDim.x + threadIdx.x;
    if (idx >= FEAT_ELEMS) return;
    int gx = (int)(idx % HW);
    size_t t = idx / HW;
    int gy = (int)(t % HW); t /= HW;
    int oc = (int)(t % NF);
    int n  = (int)(t / NF);
    float acc = hb[oc];
    for (int ic = 0; ic < 3; ++ic) {
        const float* xp = x + (((size_t)n * 3 + ic) * HW) * HW;
        float m = rgb_mean(ic);
        for (int dy = 0; dy < 3; ++dy) {
            int yy = gy + dy - 1;
            if ((unsigned)yy >= HW) continue;
            for (int dx = 0; dx < 3; ++dx) {
                int xx = gx + dx - 1;
                if ((unsigned)xx >= HW) continue;
                float v = (xp[(uint32)yy * HW + xx] - m) * KCONST;
                acc += v * hw_[((oc * 3 + ic) * 3 + dy) * 3 + dx];
            }
        }
    }
    feat[idx] = acc;
}

__global__ __launch_bounds__(256)
void resblock_kernel(const float* __restrict__ xin, float* __restrict__ xout,
                     const int8_t* __restrict__ bsign,
                     const float* __restrict__ bscale,
                     const float* __restrict__ b1, const float* __restrict__ b21,
                     const float* __restrict__ aP, const float* __restrict__ b22) {
    extern __shared__ __align__(16) int8_t act[];
    float* b1s = (float*)(act + ACT_BYTES);

    const int tid = threadIdx.x;
    const int bx0 = blockIdx.x * TW, by0 = blockIdx.y * TH;
    const int n = blockIdx.z;

    if (tid < NF) b1s[tid] = b1[tid];
    __syncthreads();

    const int NFILL = HTH * HTW * NF;
    for (int idx = tid; idx < NFILL; idx += 256) {
        int xx = idx % HTW;
        int r  = idx / HTW;
        int yy = r % HTH;
        int c  = r / HTH;
        int gx = bx0 + xx - 1, gy = by0 + yy - 1;
        int8_t s = 0;
        if ((unsigned)gx < HW && (unsigned)gy < HW) {
            float v = xin[(((uint32)n * NF + c) * HW + gy) * HW + gx] + b1s[c];
            s = (v > 0.0f) ? 1 : ((v < 0.0f) ? -1 : 0);
        }
        act[(yy * HTW + xx) * LPAD + c] = s;
    }
    __syncthreads();

    const int wave   = tid >> 5;
    const int lane   = tid & 31;
    const int mg     = wave & 3;
    const int nstrip = wave >> 2;
    const int mrow   = lane & 15;
    const int hi     = lane >> 4;

    v8i afrag[9];
#pragma unroll
    for (int t = 0; t < 9; ++t) {
        const int8_t* ap = bsign + (((size_t)(mg * 16 + mrow)) * 9 + t) * NF;
        v8i a;
#pragma unroll
        for (int v = 0; v < 8; ++v) {
            int kb = 4 * v + hi * 32;
            a[v] = *(const int*)(ap + kb);
        }
        afrag[t] = a;
    }

    float sc[8], c21[8], ca[8], c22[8];
    uint32 planeOff[8];
#pragma unroll
    for (int r = 0; r < 8; ++r) {
        int oc = mg * 16 + hi * 8 + r;
        sc[r] = bscale[oc]; c21[r] = b21[oc]; ca[r] = aP[oc]; c22[r] = b22[oc];
        planeOff[r] = (((uint32)n * NF + oc) * HW) * HW;
    }

    const int col = lane & 15;
    for (int row = nstrip * 16; row < nstrip * 16 + 16; ++row) {
        v8i acc0 = {}, acc1 = {};
#pragma unroll
        for (int t = 0; t < 9; ++t) {
            int dy = t / 3, dx = t % 3;
            const int8_t* bp0 = act + ((row + dy) * HTW + (col + dx)) * LPAD + hi * 32;
            const int8_t* bp1 = bp0 + 16 * LPAD;
            v8i b0, b1f;
#pragma unroll
            for (int q = 0; q < 4; ++q) {
                int2 d0 = *(const int2*)(bp0 + q * 8);
                int2 d1 = *(const int2*)(bp1 + q * 8);
                b0[2 * q] = d0.x; b0[2 * q + 1] = d0.y;
                b1f[2 * q] = d1.x; b1f[2 * q + 1] = d1.y;
            }
            acc0 = __builtin_amdgcn_wmma_i32_16x16x64_iu8(
                true, afrag[t], true, b0, acc0, false, false);
            acc1 = __builtin_amdgcn_wmma_i32_16x16x64_iu8(
                true, afrag[t], true, b1f, acc1, false, false);
        }

        uint32 pix0 = (uint32)(by0 + row) * HW + (uint32)(bx0 + col);
        uint32 pix1 = pix0 + 16;
#pragma unroll
        for (int r = 0; r < 8; ++r) {
            uint32 off0 = planeOff[r] + pix0;
            uint32 off1 = planeOff[r] + pix1;
            float v0 = sc[r] * (float)acc0[r] + c21[r];
            float v1 = sc[r] * (float)acc1[r] + c21[r];
            v0 = (v0 >= 0.0f) ? v0 : ca[r] * v0;
            v1 = (v1 >= 0.0f) ? v1 : ca[r] * v1;
            v0 += c22[r] + xin[off0];
            v1 += c22[r] + xin[off1];
            xout[off0] = v0;
            xout[off1] = v1;
        }
    }
}

__global__ __launch_bounds__(256)
void tail_kernel(const float* __restrict__ res, const float* __restrict__ feat,
                 const float* __restrict__ tw, const float* __restrict__ tb,
                 float* __restrict__ out) {
    __shared__ float tws[12 * NF * 9];
    for (int i = threadIdx.x; i < 12 * NF * 9; i += 256) tws[i] = tw[i];
    __syncthreads();

    size_t idx = (size_t)blockIdx.x * blockDim.x + threadIdx.x;
    if (idx >= (size_t)NIMG * HW * HW) return;
    int wx = (int)(idx % HW);
    size_t t = idx / HW;
    int wy = (int)(t % HW);
    int n  = (int)(t / HW);

    float acc[12];
#pragma unroll
    for (int o = 0; o < 12; ++o) acc[o] = tb[o];

    for (int ic = 0; ic < NF; ++ic) {
        const float* rp = res  + (((size_t)n * NF + ic) * HW) * HW;
        const float* fp = feat + (((size_t)n * NF + ic) * HW) * HW;
        for (int dy = 0; dy < 3; ++dy) {
            int yy = wy + dy - 1;
            if ((unsigned)yy >= HW) continue;
            for (int dx = 0; dx < 3; ++dx) {
                int xx = wx + dx - 1;
                if ((unsigned)xx >= HW) continue;
                float v = rp[(uint32)yy * HW + xx] + fp[(uint32)yy * HW + xx];
                const float* wp = &tws[(ic * 3 + dy) * 3 + dx];
#pragma unroll
                for (int o = 0; o < 12; ++o) acc[o] += v * wp[o * NF * 9];
            }
        }
    }
#pragma unroll
    for (int c = 0; c < 3; ++c) {
        float m = rgb_mean(c);
#pragma unroll
        for (int i = 0; i < 2; ++i)
#pragma unroll
            for (int j = 0; j < 2; ++j) {
                float v = acc[c * 4 + i * 2 + j] * (1.0f / KCONST) + m;
                int Y = wy * 2 + i, X = wx * 2 + j;
                out[(((size_t)n * 3 + c) * OUTHW + Y) * OUTHW + X] = v;
            }
    }
}

extern "C" void kernel_launch(void* const* d_in, const int* in_sizes, int n_in,
                              void* d_out, int out_size, void* d_ws, size_t ws_size,
                              hipStream_t stream) {
    const float* x      = (const float*)d_in[0];
    const float* head_w = (const float*)d_in[1];
    const float* head_b = (const float*)d_in[2];
    const float* rb_b1  = (const float*)d_in[3];
    const float* rb_w   = (const float*)d_in[4];
    const float* rb_b21 = (const float*)d_in[5];
    const float* rb_a   = (const float*)d_in[6];
    const float* rb_b22 = (const float*)d_in[7];
    const float* tail_w = (const float*)d_in[8];
    const float* tail_b = (const float*)d_in[9];
    float* out = (float*)d_out;

    char* ws = (char*)d_ws;
    float*  featF  = (float*)(ws);
    float*  bufA   = (float*)(ws + FEAT_BYTES);
    float*  bufB   = (float*)(ws + 2 * FEAT_BYTES);
    int8_t* bsign  = (int8_t*)(ws + 3 * FEAT_BYTES);
    float*  bscale = (float*)(ws + 3 * FEAT_BYTES + (size_t)NRES * NF * 9 * NF);

    prep_weights_kernel<<<(NRES * NF + 255) / 256, 256, 0, stream>>>(rb_w, bsign, bscale);

    head_kernel<<<(unsigned)(FEAT_ELEMS / 256), 256, 0, stream>>>(x, head_w, head_b, featF);

    dim3 rgrid(HW / TW, HW / TH, NIMG);
    for (int b = 0; b < NRES; ++b) {
        const float* in  = (b == 0) ? featF : ((b & 1) ? bufA : bufB);
        float*       o   = (b & 1) ? bufB : bufA;
        resblock_kernel<<<rgrid, 256, LDS_BYTES, stream>>>(
            in, o,
            bsign + (size_t)b * NF * 9 * NF, bscale + (size_t)b * NF,
            rb_b1 + (size_t)b * NF, rb_b21 + (size_t)b * NF,
            rb_a + (size_t)b * NF, rb_b22 + (size_t)b * NF);
    }
    float* resFinal = bufB;

    tail_kernel<<<(unsigned)(((size_t)NIMG * HW * HW) / 256), 256, 0, stream>>>(
        resFinal, featF, tail_w, tail_b, out);
}
